// MiniGPT_28613072126597
// MI455X (gfx1250) — matrix operand order repaired and checked by reading, not compiled, not run on hardware
//
#include <hip/hip_runtime.h>

typedef __attribute__((ext_vector_type(16))) _Float16 v16h;
typedef __attribute__((ext_vector_type(8)))  float    v8f;
typedef __attribute__((vector_size(4 * sizeof(int)))) int v4i_t;
typedef __attribute__((address_space(3))) v4i_t* lds_v4i_t;

#define V_   32000
#define T_   1024
#define D_   1024
#define H_   16
#define L_   8
#define F_   4096
#define B_   4
#define HD_  64
#define MTOT (B_ * T_)          // 4096 rows
#define EPSF 1e-5f

#if __has_builtin(__builtin_amdgcn_global_load_async_to_lds_b128)
#define USE_ASYNC 1
#else
#define USE_ASYNC 0
#endif

#if USE_ASYNC
#if __has_builtin(__builtin_amdgcn_s_wait_asynccnt)
#define WAIT_ASYNC(n) __builtin_amdgcn_s_wait_asynccnt(n)
#else
#define WAIT_ASYNC(n) asm volatile("s_wait_asynccnt " #n ::: "memory")
#endif
#else
#define WAIT_ASYNC(n) ((void)0)
#endif

union V16U { uint4 u[2]; v16h v; };

__device__ __forceinline__ v8f v8f_zero() {
    v8f z;
#pragma unroll
    for (int i = 0; i < 8; ++i) z[i] = 0.0f;
    return z;
}

__device__ __forceinline__ float gelu_exact(float x) {
    return 0.5f * x * (1.0f + erff(x * 0.70710678118654752f));
}

// ---------------------------------------------------------------------------
// Tiled transpose + fp32->f16: dst[N][K] = (f16) src[K][N]
// ---------------------------------------------------------------------------
__global__ __launch_bounds__(256) void transpose_f32_to_f16(
        const float* __restrict__ src, _Float16* __restrict__ dst, int K, int N) {
    __shared__ float tile[32][33];
    const int n0 = blockIdx.x * 32, k0 = blockIdx.y * 32;
    const int tx = threadIdx.x & 31, ty = threadIdx.x >> 5;   // ty: 0..7
#pragma unroll
    for (int i = 0; i < 4; ++i) {
        int k = ty + i * 8;
        tile[k][tx] = src[(size_t)(k0 + k) * N + n0 + tx];
    }
    __syncthreads();
#pragma unroll
    for (int i = 0; i < 4; ++i) {
        int n = ty + i * 8;
        dst[(size_t)(n0 + n) * K + k0 + tx] = (_Float16)tile[tx][n];
    }
}

// ---------------------------------------------------------------------------
// x[b,t,:] = tok_emb[idx[b,t],:] + pos_emb[t,:]
// ---------------------------------------------------------------------------
__global__ __launch_bounds__(256) void embed_kernel(const int* __restrict__ idx,
                                                    const float* __restrict__ tok,
                                                    const float* __restrict__ pos,
                                                    float* __restrict__ x) {
    size_t i = (size_t)blockIdx.x * 256 + threadIdx.x;   // over MTOT*D
    int row = (int)(i >> 10);          // b*T + t
    int d   = (int)(i & (D_ - 1));
    int t   = row & (T_ - 1);
    x[i] = tok[(size_t)idx[row] * D_ + d] + pos[(size_t)t * D_ + d];
}

// ---------------------------------------------------------------------------
// LayerNorm over D=1024, writes f16 activations. One 256-thread block per row.
// ---------------------------------------------------------------------------
__global__ __launch_bounds__(256) void layernorm_to_f16(const float* __restrict__ x,
                                                        const float* __restrict__ gamma,
                                                        const float* __restrict__ beta,
                                                        _Float16* __restrict__ out) {
    __shared__ float s1[256];
    __shared__ float s2[256];
    const int row = blockIdx.x;
    const int tid = threadIdx.x;
    const float* p = x + (size_t)row * D_;
    float4 val = *(const float4*)(p + tid * 4);
    float s = val.x + val.y + val.z + val.w;
    float q = val.x * val.x + val.y * val.y + val.z * val.z + val.w * val.w;
    s1[tid] = s; s2[tid] = q;
    __syncthreads();
    for (int off = 128; off > 0; off >>= 1) {
        if (tid < off) { s1[tid] += s1[tid + off]; s2[tid] += s2[tid + off]; }
        __syncthreads();
    }
    float mean = s1[0] * (1.0f / D_);
    float var  = s2[0] * (1.0f / D_) - mean * mean;
    float rstd = rsqrtf(var + EPSF);
    float4 gv = *(const float4*)(gamma + tid * 4);
    float4 bv = *(const float4*)(beta  + tid * 4);
    _Float16* o = out + (size_t)row * D_ + tid * 4;
    o[0] = (_Float16)((val.x - mean) * rstd * gv.x + bv.x);
    o[1] = (_Float16)((val.y - mean) * rstd * gv.y + bv.y);
    o[2] = (_Float16)((val.z - mean) * rstd * gv.z + bv.z);
    o[3] = (_Float16)((val.w - mean) * rstd * gv.w + bv.w);
}

// ---------------------------------------------------------------------------
// WMMA GEMM (TN): C[M,N] = A[M,K](f16) @ Wt[N,K](f16)^T  (+bias)(+resid)(+gelu)
// Double-buffered LDS; async global->LDS copies (ASYNCcnt) when available.
// Block tile 128x128, K-step 32, 8 waves (4x2), wave tile 32x64 (2x4 wmma).
// mode 0: outH = acc (f16)
// mode 1: outH = gelu(acc + bias) (f16)
// mode 2: outF = resid + acc + bias (f32)
// mode 3: outF = acc + bias (f32)
// mode 4: outH scattered as vT[b,h,d,t] (f16)   (V projection)
// ---------------------------------------------------------------------------
__global__ __launch_bounds__(256) void gemm_wmma_f16(const _Float16* __restrict__ A,
                                                     const _Float16* __restrict__ Wt,
                                                     const float* __restrict__ bias,
                                                     const float* __restrict__ resid,
                                                     float* __restrict__ outF,
                                                     _Float16* __restrict__ outH,
                                                     int N, int K, int mode) {
    __shared__ _Float16 As[2][128][40];    // row stride 80B (16B aligned)
    __shared__ _Float16 Ws[2][128][40];    // transposed weight tile: [n][k]

    const int tid  = threadIdx.x;
    const int lane = tid & 31;
    const int wave = tid >> 5;
    const int wm   = wave >> 1;          // 0..3 -> 32-row slice
    const int wn   = wave & 1;           // 0..1 -> 64-col slice
    const int g    = lane >> 4;          // lane group
    const int ln16 = lane & 15;
    const int mb   = blockIdx.y * 128;
    const int nb   = blockIdx.x * 128;

    v8f acc[2][4];
#pragma unroll
    for (int i = 0; i < 2; ++i)
#pragma unroll
        for (int j = 0; j < 4; ++j) acc[i][j] = v8f_zero();

    // stage one 128x32 A-tile and 128x32 W-tile into LDS buffer `buf`
    auto stage = [&](int buf, int kb) {
#if USE_ASYNC
#pragma unroll
        for (int it = 0; it < 2; ++it) {
            int s = tid + it * 256;
            int row = s >> 2, seg = s & 3;
            __builtin_amdgcn_global_load_async_to_lds_b128(
                (v4i_t*)(A + (size_t)(mb + row) * K + kb + seg * 8),
                (lds_v4i_t)&As[buf][row][seg * 8],
                0, 0);
        }
#pragma unroll
        for (int it = 0; it < 2; ++it) {
            int s = tid + it * 256;
            int row = s >> 2, seg = s & 3;
            __builtin_amdgcn_global_load_async_to_lds_b128(
                (v4i_t*)(Wt + (size_t)(nb + row) * K + kb + seg * 8),
                (lds_v4i_t)&Ws[buf][row][seg * 8],
                0, 0);
        }
#else
#pragma unroll
        for (int it = 0; it < 2; ++it) {
            int s = tid + it * 256;
            int row = s >> 2, seg = s & 3;
            uint4 va = *(const uint4*)(A  + (size_t)(mb + row) * K + kb + seg * 8);
            uint4 vw = *(const uint4*)(Wt + (size_t)(nb + row) * K + kb + seg * 8);
            *(uint4*)&As[buf][row][seg * 8] = va;
            *(uint4*)&Ws[buf][row][seg * 8] = vw;
        }
#endif
    };

    auto compute = [&](int buf) {
        // A fragments: lane group g holds K in [8g,8g+8) U [16+8g,24+8g)
        v16h af[2];
#pragma unroll
        for (int ti = 0; ti < 2; ++ti) {
            const _Float16* ap = &As[buf][wm * 32 + ti * 16 + ln16][0];
            V16U t;
            t.u[0] = *(const uint4*)(ap + 8 * g);
            t.u[1] = *(const uint4*)(ap + 16 + 8 * g);
            af[ti] = t.v;
        }
        // B fragments: lane group g holds K in [8g,8g+8) U [16+8g,24+8g)
#pragma unroll
        for (int tj = 0; tj < 4; ++tj) {
            const _Float16* bp = &Ws[buf][wn * 64 + tj * 16 + ln16][8 * g];
            V16U t;
            t.u[0] = *(const uint4*)(bp);
            t.u[1] = *(const uint4*)(bp + 16);
            v16h bf = t.v;
#pragma unroll
            for (int ti = 0; ti < 2; ++ti)
                acc[ti][tj] = __builtin_amdgcn_wmma_f32_16x16x32_f16(
                    false, af[ti], false, bf, (short)0, acc[ti][tj], false, false);
        }
    };

    stage(0, 0);
    const int nk = K >> 5;
    for (int t = 0; t < nk; ++t) {
        const int buf = t & 1;
        if (t + 1 < nk) {
            stage(buf ^ 1, (t + 1) << 5);   // prefetch next tile into other buffer
            WAIT_ASYNC(4);                  // wait: current tile resident, next in flight
        } else {
            WAIT_ASYNC(0);
        }
        __syncthreads();                    // all waves' copies complete
        compute(buf);
        __syncthreads();                    // done reading before next overwrite
    }

    // epilogue: C/D element r -> row = r + 8g, col = ln16
#pragma unroll
    for (int ti = 0; ti < 2; ++ti) {
#pragma unroll
        for (int tj = 0; tj < 4; ++tj) {
#pragma unroll
            for (int r = 0; r < 8; ++r) {
                int m = mb + wm * 32 + ti * 16 + g * 8 + r;
                int n = nb + wn * 64 + tj * 16 + ln16;
                float v = acc[ti][tj][r];
                size_t o = (size_t)m * N + n;
                if (mode == 0) {
                    outH[o] = (_Float16)v;
                } else if (mode == 1) {
                    v += bias[n];
                    outH[o] = (_Float16)gelu_exact(v);
                } else if (mode == 2) {
                    v += bias[n] + resid[o];
                    outF[o] = v;
                } else if (mode == 3) {
                    v += bias[n];
                    outF[o] = v;
                } else {
                    // mode 4: scatter V projection into vT[b,h,d,t]
                    int b = m >> 10;          // m / T
                    int t2 = m & (T_ - 1);
                    int h = n >> 6;           // n / HD
                    int d = n & (HD_ - 1);
                    outH[(((size_t)b * H_ + h) * HD_ + d) * T_ + t2] = (_Float16)v;
                }
            }
        }
    }
}

// ---------------------------------------------------------------------------
// att[bh, q, k] = scale * Q[bh,q,:] . K[bh,k,:]   (skip fully masked tiles)
// One wave per 16x16 tile; HD=64 -> two chained WMMAs.
// ---------------------------------------------------------------------------
__global__ __launch_bounds__(32) void attn_scores_wmma(const _Float16* __restrict__ q,
                                                       const _Float16* __restrict__ k,
                                                       float* __restrict__ att) {
    const int kt = blockIdx.x, qt = blockIdx.y, bh = blockIdx.z;
    if (kt > qt) return;                       // uniform: entire wave exits
    const int b = bh >> 4, h = bh & 15;
    const int lane = threadIdx.x;
    const int g = lane >> 4, ln16 = lane & 15;

    v8f c = v8f_zero();
    const _Float16* qp = q + (size_t)(b * T_ + qt * 16 + ln16) * D_ + h * HD_;
    const _Float16* kp = k + (size_t)(b * T_ + kt * 16 + ln16) * D_ + h * HD_;
#pragma unroll
    for (int s = 0; s < 2; ++s) {
        V16U ta;
        ta.u[0] = *(const uint4*)(qp + s * 32 + 8 * g);
        ta.u[1] = *(const uint4*)(qp + s * 32 + 16 + 8 * g);
        V16U tb;
        const _Float16* bp = kp + s * 32 + 8 * g;   // K in [8g,8g+8) U [16+8g,24+8g)
        tb.u[0] = *(const uint4*)(bp);
        tb.u[1] = *(const uint4*)(bp + 16);
        c = __builtin_amdgcn_wmma_f32_16x16x32_f16(false, ta.v, false, tb.v,
                                                   (short)0, c, false, false);
    }
    const float scale = 0.125f;                 // HD^-0.5
    float* out = att + (size_t)bh * T_ * T_;
#pragma unroll
    for (int r = 0; r < 8; ++r) {
        int m = qt * 16 + g * 8 + r;
        int n = kt * 16 + ln16;
        out[(size_t)m * T_ + n] = c[r] * scale;
    }
}

// ---------------------------------------------------------------------------
// Causal softmax: one wave per row of 1024 scores; writes f16 probabilities
// (zeros beyond the diagonal, so att.V can run unmasked).
// ---------------------------------------------------------------------------
__global__ __launch_bounds__(256) void softmax_causal(const float* __restrict__ att,
                                                      _Float16* __restrict__ attP) {
    const int wave = threadIdx.x >> 5, lane = threadIdx.x & 31;
    const int row  = blockIdx.x * 8 + wave;     // bh*T + qpos
    const int qpos = row & (T_ - 1);
    const float* p = att + (size_t)row * T_;
    float vals[32];
    float mx = -3.0e38f;
#pragma unroll
    for (int i = 0; i < 32; ++i) {
        int c = lane + i * 32;
        float x = (c <= qpos) ? p[c] : -3.0e38f;
        vals[i] = x;
        mx = fmaxf(mx, x);
    }
#pragma unroll
    for (int off = 16; off > 0; off >>= 1) mx = fmaxf(mx, __shfl_xor(mx, off, 32));
    float sum = 0.0f;
#pragma unroll
    for (int i = 0; i < 32; ++i) {
        int c = lane + i * 32;
        float e = (c <= qpos) ? __expf(vals[i] - mx) : 0.0f;
        vals[i] = e;
        sum += e;
    }
#pragma unroll
    for (int off = 16; off > 0; off >>= 1) sum += __shfl_xor(sum, off, 32);
    float inv = 1.0f / sum;
    _Float16* o = attP + (size_t)row * T_;
#pragma unroll
    for (int i = 0; i < 32; ++i) o[lane + i * 32] = (_Float16)(vals[i] * inv);
}

// ---------------------------------------------------------------------------
// Y[bh, q, :HD] = attP[bh, q, :] @ V[bh, :, :HD]; one wave per 16-row q tile.
// V is pre-transposed: vT[b,h,d,t] so B fragments are contiguous loads.
// K-loop truncated at the causal boundary (probs are zero past it).
// ---------------------------------------------------------------------------
__global__ __launch_bounds__(32) void attn_av_wmma(const _Float16* __restrict__ attP,
                                                   const _Float16* __restrict__ vT,
                                                   _Float16* __restrict__ y) {
    const int qt = blockIdx.x, bh = blockIdx.y;
    const int b = bh >> 4, h = bh & 15;
    const int lane = threadIdx.x;
    const int g = lane >> 4, ln16 = lane & 15;

    v8f acc[4];
#pragma unroll
    for (int j = 0; j < 4; ++j) acc[j] = v8f_zero();

    const _Float16* arow = attP + (size_t)(bh * T_ + qt * 16 + ln16) * T_;
    const int nkt = (qt >> 1) + 1;              // covers keys <= qt*16+15
    for (int kt = 0; kt < nkt; ++kt) {
        V16U ta;
        const _Float16* ap = arow + kt * 32;
        ta.u[0] = *(const uint4*)(ap + 8 * g);
        ta.u[1] = *(const uint4*)(ap + 16 + 8 * g);
#pragma unroll
        for (int tj = 0; tj < 4; ++tj) {
            int d = tj * 16 + ln16;
            const _Float16* vp = vT + ((size_t)bh * HD_ + d) * T_ + kt * 32 + 8 * g;
            V16U tb;
            tb.u[0] = *(const uint4*)(vp);
            tb.u[1] = *(const uint4*)(vp + 16);
            acc[tj] = __builtin_amdgcn_wmma_f32_16x16x32_f16(
                false, ta.v, false, tb.v, (short)0, acc[tj], false, false);
        }
    }
#pragma unroll
    for (int tj = 0; tj < 4; ++tj)
#pragma unroll
        for (int r = 0; r < 8; ++r) {
            int m = qt * 16 + g * 8 + r;
            int d = tj * 16 + ln16;
            y[(size_t)(b * T_ + m) * D_ + h * HD_ + d] = (_Float16)acc[tj][r];
        }
}

// ---------------------------------------------------------------------------
// Host-side orchestration
// ---------------------------------------------------------------------------
extern "C" void kernel_launch(void* const* d_in, const int* in_sizes, int n_in,
                              void* d_out, int out_size, void* d_ws, size_t ws_size,
                              hipStream_t stream) {
    (void)in_sizes; (void)n_in; (void)out_size; (void)ws_size;
    const int*   idx     = (const int*)  d_in[0];
    const float* tok_emb = (const float*)d_in[1];
    const float* pos_emb = (const float*)d_in[2];
    const float* wq      = (const float*)d_in[3];
    const float* wk      = (const float*)d_in[4];
    const float* wv      = (const float*)d_in[5];
    const float* wo      = (const float*)d_in[6];
    const float* bo      = (const float*)d_in[7];
    const float* ln1_g   = (const float*)d_in[8];
    const float* ln1_b   = (const float*)d_in[9];
    const float* ln2_g   = (const float*)d_in[10];
    const float* ln2_b   = (const float*)d_in[11];
    const float* w1      = (const float*)d_in[12];
    const float* b1      = (const float*)d_in[13];
    const float* w2      = (const float*)d_in[14];
    const float* b2      = (const float*)d_in[15];
    const float* lnf_g   = (const float*)d_in[16];
    const float* lnf_b   = (const float*)d_in[17];
    const float* w_head  = (const float*)d_in[18];
    const float* b_head  = (const float*)d_in[19];
    float* logits = (float*)d_out;

    char* wsp = (char*)d_ws;
    auto alloc = [&](size_t elems, size_t esz) -> char* {
        char* p = wsp;
        wsp += (elems * esz + 255) & ~(size_t)255;
        return p;
    };
    const size_t LDD = (size_t)L_ * D_ * D_;
    const size_t LDF = (size_t)L_ * D_ * F_;
    const size_t DV  = (size_t)D_ * V_;

    // all weights stored f16, TRANSPOSED: Wt[N][K]
    _Float16* wqT = (_Float16*)alloc(LDD, 2);
    _Float16* wkT = (_Float16*)alloc(LDD, 2);
    _Float16* wvT = (_Float16*)alloc(LDD, 2);
    _Float16* woT = (_Float16*)alloc(LDD, 2);
    _Float16* w1T = (_Float16*)alloc(LDF, 2);
    _Float16* w2T = (_Float16*)alloc(LDF, 2);
    _Float16* whT = (_Float16*)alloc(DV, 2);
    float*    x    = (float*)   alloc((size_t)MTOT * D_, 4);
    _Float16* h16  = (_Float16*)alloc((size_t)MTOT * D_, 2);
    _Float16* q16  = (_Float16*)alloc((size_t)MTOT * D_, 2);
    _Float16* k16  = (_Float16*)alloc((size_t)MTOT * D_, 2);
    _Float16* vT   = (_Float16*)alloc((size_t)MTOT * D_, 2);   // [b,h,d,t]
    _Float16* y16  = (_Float16*)alloc((size_t)MTOT * D_, 2);
    _Float16* mid16= (_Float16*)alloc((size_t)MTOT * F_, 2);
    float*    att  = (float*)   alloc((size_t)B_ * H_ * T_ * T_, 4);
    _Float16* attP = (_Float16*)alloc((size_t)B_ * H_ * T_ * T_, 2);

    // weight transposes (per layer; src layout [K][N] -> dst [N][K])
    for (int l = 0; l < L_; ++l) {
        size_t odd = (size_t)l * D_ * D_;
        size_t odf = (size_t)l * D_ * F_;
        transpose_f32_to_f16<<<dim3(D_ / 32, D_ / 32), 256, 0, stream>>>(wq + odd, wqT + odd, D_, D_);
        transpose_f32_to_f16<<<dim3(D_ / 32, D_ / 32), 256, 0, stream>>>(wk + odd, wkT + odd, D_, D_);
        transpose_f32_to_f16<<<dim3(D_ / 32, D_ / 32), 256, 0, stream>>>(wv + odd, wvT + odd, D_, D_);
        transpose_f32_to_f16<<<dim3(D_ / 32, D_ / 32), 256, 0, stream>>>(wo + odd, woT + odd, D_, D_);
        transpose_f32_to_f16<<<dim3(F_ / 32, D_ / 32), 256, 0, stream>>>(w1 + odf, w1T + odf, D_, F_);
        transpose_f32_to_f16<<<dim3(D_ / 32, F_ / 32), 256, 0, stream>>>(w2 + odf, w2T + odf, F_, D_);
    }
    transpose_f32_to_f16<<<dim3(V_ / 32, D_ / 32), 256, 0, stream>>>(w_head, whT, D_, V_);

    embed_kernel<<<(MTOT * D_) / 256, 256, 0, stream>>>(idx, tok_emb, pos_emb, x);

    const dim3 gdd(D_ / 128, MTOT / 128);           // 8 x 32
    const dim3 gdf(F_ / 128, MTOT / 128);           // 32 x 32
    const dim3 gdv(V_ / 128, MTOT / 128);           // 250 x 32

    for (int l = 0; l < L_; ++l) {
        size_t odd = (size_t)l * D_ * D_;
        size_t odf = (size_t)l * D_ * F_;
        layernorm_to_f16<<<MTOT, 256, 0, stream>>>(x, ln1_g + l * D_, ln1_b + l * D_, h16);
        gemm_wmma_f16<<<gdd, 256, 0, stream>>>(h16, wqT + odd, nullptr, nullptr,
                                               nullptr, q16, D_, D_, 0);
        gemm_wmma_f16<<<gdd, 256, 0, stream>>>(h16, wkT + odd, nullptr, nullptr,
                                               nullptr, k16, D_, D_, 0);
        gemm_wmma_f16<<<gdd, 256, 0, stream>>>(h16, wvT + odd, nullptr, nullptr,
                                               nullptr, vT, D_, D_, 4);
        attn_scores_wmma<<<dim3(T_ / 16, T_ / 16, B_ * H_), 32, 0, stream>>>(q16, k16, att);
        softmax_causal<<<(B_ * H_ * T_) / 8, 256, 0, stream>>>(att, attP);
        attn_av_wmma<<<dim3(T_ / 16, B_ * H_), 32, 0, stream>>>(attP, vT, y16);
        gemm_wmma_f16<<<gdd, 256, 0, stream>>>(y16, woT + odd, bo + l * D_, x,
                                               x, nullptr, D_, D_, 2);
        layernorm_to_f16<<<MTOT, 256, 0, stream>>>(x, ln2_g + l * D_, ln2_b + l * D_, h16);
        gemm_wmma_f16<<<gdf, 256, 0, stream>>>(h16, w1T + odf, b1 + l * F_, nullptr,
                                               nullptr, mid16, F_, D_, 1);
        gemm_wmma_f16<<<gdd, 256, 0, stream>>>(mid16, w2T + odf, b2 + l * D_, x,
                                               x, nullptr, D_, F_, 2);
    }
    layernorm_to_f16<<<MTOT, 256, 0, stream>>>(x, lnf_g, lnf_b, h16);
    gemm_wmma_f16<<<gdv, 256, 0, stream>>>(h16, whT, b_head, nullptr,
                                           logits, nullptr, V_, D_, 3);
}
